// MultiRegressionLSTM_44487271252614
// MI455X (gfx1250) — compile-verified, matrix operand order checked by reading, not run on hardware
//
#include <hip/hip_runtime.h>
#include <hip/hip_bf16.h>

// ---------------------------------------------------------------------------
// 2-layer LSTM + FC, persistent-workgroup design for MI455X (gfx1250).
//   B=2048, T=64, I=128, H=1024.  ~3.5 TFLOP, weights L2-resident -> compute
//   bound -> bf16 WMMA (v_wmma_f32_16x16x32_bf16).
// Each block owns 16 batch rows for all 64 timesteps (no global sync).
// Weights are packed once per call into d_ws as bf16 in WMMA B-operand layout.
// v3: 512 threads / 16 waves (4 n-tiles per wave) to cut VGPRs/wave (~<=256)
//     and raise waves/SIMD; B tiles staged in locals so the 16 global_load_b128
//     issue as one clause with staggered s_wait_loadcnt before the 8 wmmas.
// ---------------------------------------------------------------------------

typedef __attribute__((ext_vector_type(16))) __bf16 v16bf;
typedef __attribute__((ext_vector_type(8)))  float  v8f;

constexpr int B_ = 2048;
constexpr int T_ = 64;
constexpr int I_ = 128;
constexpr int H_ = 1024;
constexpr int G_ = 4 * H_;   // 4096 gate rows

constexpr int MT  = 16;           // batch rows per block
constexpr int NTH = 512;          // threads per block (16 waves)
constexpr int NTW = 4;            // n-tiles per wave (64 tiles / 16 waves)
constexpr int HP  = H_ + 16;      // padded LDS row stride (bf16 elems)
constexpr int IP  = I_ + 16;

// d_out layout: output_seq [B,T,I] ++ h1 ++ c1 ++ h2 ++ c2 (each [B,H]), f32
constexpr size_t O_H1 = (size_t)B_ * T_ * I_;
constexpr size_t O_C1 = O_H1 + (size_t)B_ * H_;
constexpr size_t O_H2 = O_C1 + (size_t)B_ * H_;
constexpr size_t O_C2 = O_H2 + (size_t)B_ * H_;

// ---------------- small helpers ----------------
__device__ __forceinline__ unsigned short bf16bits(float f) {
  union { float f; unsigned u; } x; x.f = f;
  unsigned r = x.u + 0x7FFFu + ((x.u >> 16) & 1u);   // round-to-nearest-even
  return (unsigned short)(r >> 16);
}
__device__ __forceinline__ unsigned packbf(float lo, float hi) {
  return (unsigned)bf16bits(lo) | ((unsigned)bf16bits(hi) << 16);
}
__device__ __forceinline__ float bf2f(unsigned short s) {
  union { unsigned u; float f; } y; y.u = ((unsigned)s) << 16; return y.f;
}
__device__ __forceinline__ float sig(float v) { return 1.f / (1.f + __expf(-v)); }
__device__ __forceinline__ float tnh(float v) {
  float e = __expf(-2.f * v);
  return (1.f - e) / (1.f + e);
}
__device__ __forceinline__ v8f zero8() {
  v8f z = {0.f, 0.f, 0.f, 0.f, 0.f, 0.f, 0.f, 0.f};
  return z;
}
__device__ __forceinline__ v8f wmma_bf16(v16bf a, v16bf b, v8f c) {
  return __builtin_amdgcn_wmma_f32_16x16x32_bf16(
      /*neg_a=*/false, a, /*neg_b=*/false, b,
      /*c_mod=*/(short)0, c, /*reuse_a=*/false, /*reuse_b=*/false);
}

// A-tile (16x32 bf16) fragment load; q = row base (already lane-adjusted),
// two 16B halves at +0 and +16 elems (K-half split per ISA 16-bit A layout).
__device__ __forceinline__ v16bf load_A(const unsigned short* q) {
  union { uint4 u[2]; v16bf v; } t;
  t.u[0] = *(const uint4*)(q);
  t.u[1] = *(const uint4*)(q + 16);
  return t.v;
}

// ---------------- weight pre-pack (one pass, bf16, WMMA B layout) -----------
// src: row-major W[J][K] (gates = act @ W^T).  Packed as tiles of 512 bf16:
// tile(jt,kt) holds B-block K=kt*32..+32, N=jt*16..+16; within the tile lane l
// stores 16 contiguous bf16: n = l&15, K-halves split by l>>4 (mirrors A).
__global__ __launch_bounds__(256) void pack_wmma_b(const float* __restrict__ src,
                                                   unsigned short* __restrict__ dst,
                                                   int K, long total) {
  long idx = (long)blockIdx.x * 256 + threadIdx.x;
  if (idx >= total) return;
  int  within = (int)(idx & 511);
  long tile   = idx >> 9;
  int  lane = within >> 4;
  int  s    = within & 15;
  int  ktiles = K >> 5;
  long jt = tile / ktiles;
  int  kt = (int)(tile - jt * ktiles);
  int  n = lane & 15, grp = lane >> 4;
  int  k = kt * 32 + (s & 7) + ((s & 8) ? 16 : 0) + (grp ? 8 : 0);
  long j = jt * 16 + n;
  dst[idx] = bf16bits(src[j * (long)K + k]);
}

__global__ __launch_bounds__(256) void bias_add(const float* __restrict__ a,
                                                const float* __restrict__ b,
                                                float* __restrict__ o, int n) {
  int i = blockIdx.x * 256 + threadIdx.x;
  if (i < n) o[i] = a[i] + b[i];
}

// ---------------- one LSTM layer step for this block's 16-row tile ----------
// KT1 = #K-tiles of the first GEMM input (x: 4, h1: 32); recurrent K is 32.
// Each wave owns 4 of the 64 H column-tiles, processed in 2 groups of 2 so the
// group index is compile-time (c-state stays in registers).  K-loops stay
// rolled; all 8 B tiles of an iteration are staged in locals first so the
// backend issues one deep load clause and staggers the loadcnt waits.
template <int KT1>
__device__ __forceinline__ void lstm_layer(
    const unsigned short* in1, int rs1, const unsigned short* __restrict__ wp1,
    const unsigned short* hprev, const unsigned short* __restrict__ wp2,
    const float* __restrict__ bias, v8f* ct, unsigned short* hout,
    int wave, int lane) {
  const int gr   = lane >> 4;
  const int n16  = lane & 15;
  const int boff = lane << 4;        // 16 bf16 per lane inside a 512-elem tile
  const unsigned short* a1row = in1 + n16 * rs1 + gr * 8;
  const unsigned short* a2row = hprev + n16 * HP + gr * 8;
#pragma unroll
  for (int grp = 0; grp < 2; ++grp) {
    const int nt0 = wave * NTW + grp * 2;
    v8f acc[2][4];
#pragma unroll
    for (int jj = 0; jj < 2; ++jj)
#pragma unroll
      for (int g = 0; g < 4; ++g) acc[jj][g] = zero8();

    {   // first input GEMM (x or h1), K = KT1*32
      const unsigned short* b1[2][4];
#pragma unroll
      for (int jj = 0; jj < 2; ++jj)
#pragma unroll
        for (int g = 0; g < 4; ++g)
          b1[jj][g] = wp1 + (((size_t)(g * 64 + nt0 + jj) * KT1) << 9) + boff;
#pragma unroll 1
      for (int kt = 0; kt < KT1; ++kt) {
        v16bf a = load_A(a1row + kt * 32);
        v16bf bt[2][4];
#pragma unroll
        for (int jj = 0; jj < 2; ++jj)
#pragma unroll
          for (int g = 0; g < 4; ++g)
            bt[jj][g] = *(const v16bf*)(b1[jj][g] + ((size_t)kt << 9));
#pragma unroll
        for (int jj = 0; jj < 2; ++jj)
#pragma unroll
          for (int g = 0; g < 4; ++g)
            acc[jj][g] = wmma_bf16(a, bt[jj][g], acc[jj][g]);
      }
    }
    {   // recurrent GEMM, K = 1024
      const unsigned short* b2[2][4];
#pragma unroll
      for (int jj = 0; jj < 2; ++jj)
#pragma unroll
        for (int g = 0; g < 4; ++g)
          b2[jj][g] = wp2 + (((size_t)(g * 64 + nt0 + jj) * 32) << 9) + boff;
#pragma unroll 1
      for (int kt = 0; kt < 32; ++kt) {
        v16bf a = load_A(a2row + kt * 32);
        v16bf bt[2][4];
#pragma unroll
        for (int jj = 0; jj < 2; ++jj)
#pragma unroll
          for (int g = 0; g < 4; ++g)
            bt[jj][g] = *(const v16bf*)(b2[jj][g] + ((size_t)kt << 9));
#pragma unroll
        for (int jj = 0; jj < 2; ++jj)
#pragma unroll
          for (int g = 0; g < 4; ++g)
            acc[jj][g] = wmma_bf16(a, bt[jj][g], acc[jj][g]);
      }
    }
    // bias + activations + cell update + h -> LDS (bf16)
#pragma unroll
    for (int jj = 0; jj < 2; ++jj) {
      const int j  = grp * 2 + jj;
      const int c0 = (wave * NTW + j) * 16;
      const float bi = bias[c0 + n16];
      const float bf = bias[H_ + c0 + n16];
      const float bg = bias[2 * H_ + c0 + n16];
      const float bo = bias[3 * H_ + c0 + n16];
#pragma unroll
      for (int r = 0; r < 8; ++r) {
        const float iv = sig(acc[jj][0][r] + bi);
        const float fv = sig(acc[jj][1][r] + bf);
        const float gv = tnh(acc[jj][2][r] + bg);
        const float ov = sig(acc[jj][3][r] + bo);
        const float cn = fv * ct[j][r] + iv * gv;
        ct[j][r] = cn;
        hout[(r + 8 * gr) * HP + c0 + n16] = bf16bits(ov * tnh(cn));
      }
    }
  }
}

// ---------------- persistent main kernel ------------------------------------
__global__ __launch_bounds__(NTH, 1) void lstm_persistent(
    const float* __restrict__ x,
    const float* __restrict__ h1_in, const float* __restrict__ c1_in,
    const float* __restrict__ h2_in, const float* __restrict__ c2_in,
    const unsigned short* __restrict__ wih1, const unsigned short* __restrict__ whh1,
    const unsigned short* __restrict__ wih2, const unsigned short* __restrict__ whh2,
    const unsigned short* __restrict__ fcw,
    const float* __restrict__ bias1, const float* __restrict__ bias2,
    const float* __restrict__ fc_b,
    float* __restrict__ out) {
  __shared__ unsigned short xs[MT * IP];
  __shared__ unsigned short h1s[2][MT * HP];
  __shared__ unsigned short h2s[2][MT * HP];

  const int tid  = threadIdx.x;
  const int lane = tid & 31;
  const int wave = tid >> 5;
  const int gr   = lane >> 4;
  const int n16  = lane & 15;
  const int r0   = blockIdx.x * MT;

  // persistent cell state: 4 column tiles x 8 rows per lane, per layer
  v8f c1t[NTW], c2t[NTW];
#pragma unroll
  for (int j = 0; j < NTW; ++j) {
    const int c0 = (wave * NTW + j) * 16;
#pragma unroll
    for (int r = 0; r < 8; ++r) {
      const int m = r + 8 * gr;
      c1t[j][r] = c1_in[(size_t)(r0 + m) * H_ + c0 + n16];
      c2t[j][r] = c2_in[(size_t)(r0 + m) * H_ + c0 + n16];
    }
  }
  // initial h -> ping-pong buffer 1 (read at t=0); packed 2-element stores
#pragma unroll 1
  for (int i = tid; i < MT * (H_ / 2); i += NTH) {
    const int r = i >> 9;
    const int c = (i & 511) * 2;
    const size_t g = (size_t)(r0 + r) * H_ + c;
    *(unsigned*)&h1s[1][r * HP + c] = packbf(h1_in[g], h1_in[g + 1]);
    *(unsigned*)&h2s[1][r * HP + c] = packbf(h2_in[g], h2_in[g + 1]);
  }
  __syncthreads();

#pragma unroll 1
  for (int t = 0; t < T_; ++t) {
    const int cur = t & 1, prv = cur ^ 1;
    {   // stage x_t tile -> LDS bf16: 4 packed bf16 per thread (b64 store)
      const int row = tid >> 5;
      const int cc  = (tid & 31) * 4;
      const float* xp = x + (size_t)(r0 + row) * T_ * I_ + (size_t)t * I_ + cc;
      float4 f0 = *(const float4*)xp;
      uint2 pk;
      pk.x = packbf(f0.x, f0.y);
      pk.y = packbf(f0.z, f0.w);
      *(uint2*)&xs[row * IP + cc] = pk;
    }
    __syncthreads();
    lstm_layer<4>(xs, IP, wih1, h1s[prv], whh1, bias1, c1t, h1s[cur], wave, lane);
    __syncthreads();
    lstm_layer<32>(h1s[cur], HP, wih2, h2s[prv], whh2, bias2, c2t, h2s[cur], wave, lane);
    __syncthreads();
    if (wave < 8) {  // FC: y_t[16,128] = h2 @ fc_w^T + fc_b; wave w -> cols w*16..+16
      const int c0 = wave * 16;
      const unsigned short* a2row = h2s[cur] + n16 * HP + gr * 8;
      const unsigned short* bb2 = fcw + (((size_t)wave * 32) << 9) + (lane << 4);
      v8f acc = zero8();
#pragma unroll 1
      for (int kt = 0; kt < 32; ++kt) {
        v16bf a = load_A(a2row + kt * 32);
        v16bf b = *(const v16bf*)(bb2 + ((size_t)kt << 9));
        acc = wmma_bf16(a, b, acc);
      }
      const float bb = fc_b[c0 + n16];
#pragma unroll
      for (int r = 0; r < 8; ++r) {
        const int m = r + 8 * gr;
        out[(size_t)(r0 + m) * T_ * I_ + (size_t)t * I_ + c0 + n16] = acc[r] + bb;
      }
    }
  }
  __syncthreads();

  // final states: h from LDS buffer (T-1)&1 == 1, c from registers
#pragma unroll 1
  for (int i = tid; i < MT * H_; i += NTH) {
    const int r = i >> 10, c = i & (H_ - 1);
    out[O_H1 + (size_t)(r0 + r) * H_ + c] = bf2f(h1s[1][r * HP + c]);
    out[O_H2 + (size_t)(r0 + r) * H_ + c] = bf2f(h2s[1][r * HP + c]);
  }
#pragma unroll
  for (int j = 0; j < NTW; ++j) {
    const int c0 = (wave * NTW + j) * 16;
#pragma unroll
    for (int r = 0; r < 8; ++r) {
      const int m = r + 8 * gr;
      out[O_C1 + (size_t)(r0 + m) * H_ + c0 + n16] = c1t[j][r];
      out[O_C2 + (size_t)(r0 + m) * H_ + c0 + n16] = c2t[j][r];
    }
  }
}

// ---------------- launch -----------------------------------------------------
extern "C" void kernel_launch(void* const* d_in, const int* in_sizes, int n_in,
                              void* d_out, int out_size, void* d_ws, size_t ws_size,
                              hipStream_t stream) {
  const float* x     = (const float*)d_in[0];
  const float* h1_in = (const float*)d_in[1];
  const float* c1_in = (const float*)d_in[2];
  const float* h2_in = (const float*)d_in[3];
  const float* c2_in = (const float*)d_in[4];
  const float* w_ih1 = (const float*)d_in[5];
  const float* w_hh1 = (const float*)d_in[6];
  const float* b_ih1 = (const float*)d_in[7];
  const float* b_hh1 = (const float*)d_in[8];
  const float* w_ih2 = (const float*)d_in[9];
  const float* w_hh2 = (const float*)d_in[10];
  const float* b_ih2 = (const float*)d_in[11];
  const float* b_hh2 = (const float*)d_in[12];
  const float* fc_w  = (const float*)d_in[13];
  const float* fc_b  = (const float*)d_in[14];
  float* out = (float*)d_out;

  // workspace layout (bf16 packed weights then f32 summed biases), ~26.5 MB
  unsigned short* p_wih1 = (unsigned short*)d_ws;
  unsigned short* p_whh1 = p_wih1 + (size_t)G_ * I_;
  unsigned short* p_wih2 = p_whh1 + (size_t)G_ * H_;
  unsigned short* p_whh2 = p_wih2 + (size_t)G_ * H_;
  unsigned short* p_fcw  = p_whh2 + (size_t)G_ * H_;
  float* p_b1 = (float*)(p_fcw + (size_t)I_ * H_);
  float* p_b2 = p_b1 + G_;

  auto blocks = [](long n) { return (unsigned)((n + 255) / 256); };

  long n1 = (long)G_ * I_;   // w_ih1
  long nh = (long)G_ * H_;   // w_hh1 / w_ih2 / w_hh2
  long nf = (long)I_ * H_;   // fc_w
  pack_wmma_b<<<blocks(n1), 256, 0, stream>>>(w_ih1, p_wih1, I_, n1);
  pack_wmma_b<<<blocks(nh), 256, 0, stream>>>(w_hh1, p_whh1, H_, nh);
  pack_wmma_b<<<blocks(nh), 256, 0, stream>>>(w_ih2, p_wih2, H_, nh);
  pack_wmma_b<<<blocks(nh), 256, 0, stream>>>(w_hh2, p_whh2, H_, nh);
  pack_wmma_b<<<blocks(nf), 256, 0, stream>>>(fc_w,  p_fcw,  H_, nf);
  bias_add<<<blocks(G_), 256, 0, stream>>>(b_ih1, b_hh1, p_b1, G_);
  bias_add<<<blocks(G_), 256, 0, stream>>>(b_ih2, b_hh2, p_b2, G_);

  lstm_persistent<<<B_ / MT, NTH, 0, stream>>>(
      x, h1_in, c1_in, h2_in, c2_in,
      p_wih1, p_whh1, p_wih2, p_whh2, p_fcw,
      p_b1, p_b2, fc_b, out);
}
